// KVAEModel_61014305407695
// MI455X (gfx1250) — matrix operand order repaired and checked by reading, not compiled, not run on hardware
//
#include <hip/hip_runtime.h>

// ---------------- constants ----------------
constexpr int SEQ = 128, BB = 32, SBN = SEQ * BB;   // 4096 images
constexpr int XD = 1024, AD = 8, ZD = 4, KMIX = 3, HH = 50, CN = 288;

typedef __attribute__((ext_vector_type(16))) _Float16     half16;
typedef __attribute__((ext_vector_type(8)))  float        floatx8;
typedef __attribute__((ext_vector_type(4)))  unsigned int uint4r;
typedef __attribute__((ext_vector_type(4)))  unsigned int uint32x4;
typedef __attribute__((ext_vector_type(4)))  int          int32x4;
typedef __attribute__((ext_vector_type(8)))  int          int32x8;

__device__ __forceinline__ float sigmoidf_(float x) { return 1.f / (1.f + __expf(-x)); }

// ---------------- small conversion / packing kernels ----------------
__global__ void f32_to_f16_kernel(const float* __restrict__ src, _Float16* __restrict__ dst, int n) {
    int i = blockIdx.x * blockDim.x + threadIdx.x;
    if (i < n) dst[i] = (_Float16)src[i];
}

// Pack conv weights directly into WMMA-B *fragment* layout:
//   element index = ((nt*nKb + kb)*32 + lane)*16 + i
//   n = nt*16 + (lane&15) ; k = kb*32 + ((lane>>4)<<3) + (i&7) + ((i>>3)<<4) ; k = ci*9 + ky*3 + kx
// transposed==0: W is (Co,Ci,3,3).  transposed==1 (convT): W is (Ci,Co,3,3), spatially flipped.
__global__ void pack_conv_w(const float* __restrict__ W, _Float16* __restrict__ Bp,
                            int Ci, int Co, int Kp, int transposed) {
    int idx = blockIdx.x * blockDim.x + threadIdx.x;
    if (idx >= Kp * Co) return;
    int i = idx & 15, lane = (idx >> 4) & 31, fb = idx >> 9;
    int nKb = Kp >> 5;
    int kb = fb % nKb, nt = fb / nKb;
    int n = nt * 16 + (lane & 15);
    int k = kb * 32 + ((lane >> 4) << 3) + (i & 7) + ((i >> 3) << 4);
    _Float16 v = (_Float16)0.f;
    if (k < Ci * 9 && n < Co) {
        int ci = k / 9, kr = k - ci * 9, ky = kr / 3, kx = kr - ky * 3;
        float w = transposed ? W[((ci * Co + n) * 3 + (2 - ky)) * 3 + (2 - kx)]
                             : W[((n * Ci + ci) * 3 + ky) * 3 + kx];
        v = (_Float16)w;
    }
    Bp[idx] = v;
}

// Pack linear weight W (N, Kreal) into fragment layout (Kp x Np logical), zero-padded.
__global__ void pack_mat_T(const float* __restrict__ W, _Float16* __restrict__ Bp,
                           int N, int Kreal, int Kp, int Np) {
    int idx = blockIdx.x * blockDim.x + threadIdx.x;
    if (idx >= Kp * Np) return;
    int i = idx & 15, lane = (idx >> 4) & 31, fb = idx >> 9;
    int nKb = Kp >> 5;
    int kb = fb % nKb, nt = fb / nKb;
    int n = nt * 16 + (lane & 15);
    int k = kb * 32 + ((lane >> 4) << 3) + (i & 7) + ((i >> 3) << 4);
    _Float16 v = (_Float16)0.f;
    if (k < Kreal && n < N) v = (_Float16)W[n * Kreal + k];
    Bp[idx] = v;
}

// ---------------- WMMA A-fragment builders (wave32 layout, ISA 7.12.2) ----------------
// Gather 8 consecutive K elements with an incrementally-carried (ci,kr) decomposition:
// no divisions in the hot loop; ky = (kr>=3)+(kr>=6), kx = kr - 3*ky.
// mode 0 (stride-2 VALID conv): rowBase = img*Ci*HiWi + 2*oy*Wi + 2*ox  (always in-bounds)
// mode 1 (convT pad2/dil2):     rowBase = img*Ci*HiWi ; pyb = oy-2 ; pxb = ox-2
__device__ __forceinline__ void gather8(_Float16* __restrict__ dst, const _Float16* __restrict__ inp,
        bool mv, size_t rowBase, int pyb, int pxb, int Hi, int Wi, int HiWi,
        int k, int Kreal, int ci, int kr, int mode) {
    for (int i = 0; i < 8; ++i) {
        _Float16 v = (_Float16)0.f;
        if (mv && k < Kreal) {
            int ky = (kr >= 3) + (kr >= 6);
            int kx = kr - ky * 3;
            if (mode == 0) {
                v = inp[rowBase + (size_t)ci * HiWi + ky * Wi + kx];
            } else {
                int py = pyb + ky, px = pxb + kx;
                bool ok = ((py | px) >= 0) && (((py & 1) | (px & 1)) == 0);
                int iy = py >> 1, ix = px >> 1;
                if (ok && iy < Hi && ix < Wi)
                    v = inp[rowBase + (size_t)ci * HiWi + iy * Wi + ix];
            }
        }
        dst[i] = v;
        ++k; ++kr; if (kr == 9) { kr = 0; ++ci; }
    }
}

// linear mode: row = inp + m*Kreal (contiguous, 16B-aligned vector loads when fully valid)
__device__ __forceinline__ half16 load_a_lin(const _Float16* __restrict__ row,
                                             bool mv, int lane, int kk, int Kreal) {
    union { half16 v; _Float16 h[16]; uint4r q[2]; } a;
    int khalf = (lane >> 4) * 8;
    for (int half = 0; half < 2; ++half) {
        int base = kk + khalf + half * 16;
        if (mv && base + 8 <= Kreal) {
            a.q[half] = *(const uint4r*)(row + base);
        } else {
            for (int i = 0; i < 8; ++i) {
                int k = base + i;
                a.h[half * 8 + i] = (mv && k < Kreal) ? row[k] : (_Float16)0.f;
            }
        }
    }
    return a.v;
}

// ---------------- epilogue helper ----------------
__device__ __forceinline__ void store_tile(floatx8 acc, int nbase, int lane, int m0,
        int Mtot, int hw, int Ho, int Wo, int Co, int mode, int relu_flag,
        const float* __restrict__ bias, _Float16* __restrict__ out16, float* __restrict__ out32) {
    union { floatx8 v; float f[8]; } c; c.v = acc;
    int n = nbase + (lane & 15);
    int mb = (lane >> 4) * 8;
    float bb = bias ? bias[n] : 0.f;
    for (int r = 0; r < 8; ++r) {
        int m = m0 + mb + r;
        if (m < Mtot) {
            float v = c.f[r] + bb;
            if (relu_flag) v = fmaxf(v, 0.f);
            size_t off;
            if (mode == 2) off = (size_t)m * Co + n;
            else {
                int img = m / hw, rm = m - img * hw, oy = rm / Wo, ox = rm - oy * Wo;
                off = (((size_t)img * Co + n) * Ho + oy) * Wo + ox;
            }
            if (out16) out16[off] = (_Float16)v;
            if (out32) out32[off] = v;
        }
    }
}

// ---------------- generic conv / convT / linear GEMM via WMMA ----------------
// mode 0: stride-2 VALID conv.  mode 1: transposed conv.  mode 2: linear (inp is (M,Kreal)).
// Weights Bw are in fragment layout; staged to dynamic LDS via the Tensor Data Mover.
// If Wg != nullptr: fuse 1x1 gate conv (channel reduce over Co=32) + sigmoid -> out32 (Nimg, Ho*Wo).
__global__ void conv_gemm(const _Float16* __restrict__ inp, const _Float16* __restrict__ Bw,
                          const float* __restrict__ bias,
                          _Float16* __restrict__ out16, float* __restrict__ out32,
                          const float* __restrict__ Wg, const float* __restrict__ bgp,
                          int Nimg, int Ci, int Hi, int Wi, int Ho, int Wo, int Co,
                          int Kreal, int Kp, int mode, int relu_flag) {
    extern __shared__ half16 sBv[];          // (Co/16)*(Kp/32)*32 fragments, dynamic LDS @ offset 0
    int tid = threadIdx.x;
    int nKb = Kp >> 5;
    int nFrag = (Co >> 4) * nKb * 32;

#if __has_builtin(__builtin_amdgcn_tensor_load_to_lds) && __has_builtin(__builtin_amdgcn_s_wait_tensorcnt)
    if (tid < 32) {
        // Tensor DMA: flat copy of N f16 elements (one row) global -> LDS offset 0.
        unsigned long long ga = (unsigned long long)Bw;
        unsigned N = (unsigned)(Kp * Co);
        // D# group0: count=1 | lds_addr=0 | global_addr | type=2   (ISA 8.3)
        uint32x4 g0 = { 1u, 0u, (unsigned)ga,
                        (unsigned)((ga >> 32) & 0x01ffffffu) | 0x80000000u };
        // D# group1: data_size=2B; tensor_dim0=N; tensor_dim1=1; tile_dim0=N; tile_dim1=1;
        //            tensor_dim0_stride=N (ISA 8.4)
        int32x8 g1 = { (int)0x00010000u,
                       (int)((N & 0xffffu) << 16),
                       (int)((N >> 16) | (1u << 16)),
                       (int)((N & 0xffffu) << 16),
                       1,
                       (int)N,
                       0, 0 };
        int32x4 z4 = { 0, 0, 0, 0 };
#if __clang_major__ >= 23
        int32x8 z8 = { 0, 0, 0, 0, 0, 0, 0, 0 };
        __builtin_amdgcn_tensor_load_to_lds(g0, g1, z4, z4, z8, 0);
#else
        __builtin_amdgcn_tensor_load_to_lds(g0, g1, z4, z4, 0);
#endif
        __builtin_amdgcn_s_wait_tensorcnt(0);
    }
#else
    {
        const half16* src = (const half16*)Bw;
        for (int i = tid; i < nFrag; i += blockDim.x) sBv[i] = src[i];
    }
#endif
    __syncthreads();

    int lane = tid & 31, wave = tid >> 5;
    int hw = Ho * Wo;
    int HiWi = Hi * Wi;
    int Mtot = Nimg * hw;
    int m0 = (blockIdx.x * (blockDim.x >> 5) + wave) * 16;
    if (m0 >= Mtot) return;

    // per-wave row decomposition (loop-invariant over K)
    int mrow = m0 + (lane & 15);
    bool mv = (mrow < Mtot);
    int img = 0, oy = 0, ox = 0;
    if (mv) { img = mrow / hw; int r = mrow - img * hw; oy = r / Wo; ox = r - oy * Wo; }
    size_t rowBase = (size_t)img * Ci * HiWi;
    int pyb = oy - 2, pxb = ox - 2;
    if (mode == 0) rowBase += (size_t)(2 * oy) * Wi + 2 * ox;
    const _Float16* lin_row = inp + (size_t)mrow * Kreal;

    // per-half K decomposition, carried across the K loop (advance by 32: kr += 5 mod 9, ci += 3)
    int khalf = (lane >> 4) * 8;
    int baseA = khalf, baseB = khalf + 16;
    int ciA = baseA / 9, krA = baseA - ciA * 9;
    int ciB = baseB / 9, krB = baseB - ciB * 9;

    if (Co == 32) {
        floatx8 acc0 = {0,0,0,0,0,0,0,0}, acc1 = {0,0,0,0,0,0,0,0};
        for (int kb = 0; kb < nKb; ++kb) {
            union { half16 v; _Float16 h[16]; } a;
            gather8(a.h,     inp, mv, rowBase, pyb, pxb, Hi, Wi, HiWi, baseA, Kreal, ciA, krA, mode);
            gather8(a.h + 8, inp, mv, rowBase, pyb, pxb, Hi, Wi, HiWi, baseB, Kreal, ciB, krB, mode);
            half16 b0 = sBv[kb * 32 + lane];
            half16 b1 = sBv[(nKb + kb) * 32 + lane];
            acc0 = __builtin_amdgcn_wmma_f32_16x16x32_f16(false, a.v, false, b0, (short)0, acc0, false, false);
            acc1 = __builtin_amdgcn_wmma_f32_16x16x32_f16(false, a.v, false, b1, (short)0, acc1, false, false);
            baseA += 32; krA += 5; if (krA >= 9) { krA -= 9; ++ciA; } ciA += 3;
            baseB += 32; krB += 5; if (krB >= 9) { krB -= 9; ++ciB; } ciB += 3;
        }
        if (Wg != nullptr) {
            // fused convT2 -> relu -> 1x1 gate -> sigmoid
            union { floatx8 v; float f[8]; } c0, c1; c0.v = acc0; c1.v = acc1;
            int n = lane & 15, mb = (lane >> 4) * 8;
            float w0 = Wg[n], w1 = Wg[16 + n], bb0 = bias[n], bb1 = bias[16 + n], bg = bgp[0];
            for (int r = 0; r < 8; ++r) {
                float v = fmaxf(c0.f[r] + bb0, 0.f) * w0 + fmaxf(c1.f[r] + bb1, 0.f) * w1;
                for (int off = 1; off < 16; off <<= 1) v += __shfl_xor(v, off, 32);
                int m = m0 + mb + r;
                if ((lane & 15) == 0 && m < Mtot) {
                    int im2 = m / hw, rm = m - im2 * hw;
                    out32[(size_t)im2 * hw + rm] = sigmoidf_(v + bg);
                }
            }
        } else {
            store_tile(acc0, 0, lane, m0, Mtot, hw, Ho, Wo, Co, mode, relu_flag, bias, out16, out32);
            store_tile(acc1, 16, lane, m0, Mtot, hw, Ho, Wo, Co, mode, relu_flag, bias, out16, out32);
        }
    } else {
        int ntiles = Co >> 4;
        for (int nt = 0; nt < ntiles; ++nt) {
            floatx8 acc = {0,0,0,0,0,0,0,0};
            int bA = baseA, bB = baseB, cA = ciA, rA = krA, cB = ciB, rB = krB;
            for (int kb = 0; kb < nKb; ++kb) {
                union { half16 v; _Float16 h[16]; } a;
                if (mode == 2) {
                    a.v = load_a_lin(lin_row, mv, lane, kb * 32, Kreal);
                } else {
                    gather8(a.h,     inp, mv, rowBase, pyb, pxb, Hi, Wi, HiWi, bA, Kreal, cA, rA, mode);
                    gather8(a.h + 8, inp, mv, rowBase, pyb, pxb, Hi, Wi, HiWi, bB, Kreal, cB, rB, mode);
                    bA += 32; rA += 5; if (rA >= 9) { rA -= 9; ++cA; } cA += 3;
                    bB += 32; rB += 5; if (rB >= 9) { rB -= 9; ++cB; } cB += 3;
                }
                half16 b = sBv[(nt * nKb + kb) * 32 + lane];
                acc = __builtin_amdgcn_wmma_f32_16x16x32_f16(false, a.v, false, b, (short)0, acc, false, false);
            }
            store_tile(acc, nt * 16, lane, m0, Mtot, hw, Ho, Wo, Co, mode, relu_flag, bias, out16, out32);
        }
    }
}

// ---------------- encoder head: feat -> a_mean, a_logvar, a (fused reparam) via WMMA ----------------
__global__ void encode_head(const _Float16* __restrict__ feat,
                            const _Float16* __restrict__ pWm, const _Float16* __restrict__ pWlv,
                            const float* __restrict__ bm, const float* __restrict__ blv,
                            const float* __restrict__ eps,
                            float* __restrict__ out_a, float* __restrict__ out_mean,
                            float* __restrict__ out_lv, _Float16* __restrict__ a16) {
    __shared__ half16 sBm[9 * 32];
    __shared__ half16 sBl[9 * 32];
    int tid = threadIdx.x;
    for (int i = tid; i < 9 * 32; i += blockDim.x) {
        sBm[i] = ((const half16*)pWm)[i];
        sBl[i] = ((const half16*)pWlv)[i];
    }
    __syncthreads();
    int lane = tid & 31, wave = tid >> 5;
    int m0 = (blockIdx.x * (blockDim.x >> 5) + wave) * 16;
    if (m0 >= SBN) return;
    int mrow = m0 + (lane & 15);
    bool mv = (mrow < SBN);
    const _Float16* row = feat + (size_t)mrow * CN;
    floatx8 am = {0,0,0,0,0,0,0,0}, al = {0,0,0,0,0,0,0,0};
    for (int kb = 0; kb < 9; ++kb) {
        half16 a  = load_a_lin(row, mv, lane, kb * 32, CN);
        half16 b0 = sBm[kb * 32 + lane];
        half16 b1 = sBl[kb * 32 + lane];
        am = __builtin_amdgcn_wmma_f32_16x16x32_f16(false, a, false, b0, (short)0, am, false, false);
        al = __builtin_amdgcn_wmma_f32_16x16x32_f16(false, a, false, b1, (short)0, al, false, false);
    }
    union { floatx8 v; float f[8]; } cm, cl; cm.v = am; cl.v = al;
    int n = lane & 15, mb = (lane >> 4) * 8;
    if (n < AD) {
        for (int r = 0; r < 8; ++r) {
            int m = m0 + mb + r;
            if (m < SBN) {
                float mean = cm.f[r] + bm[n];
                float lv = sigmoidf_(cl.f[r] + blv[n]);
                float av = mean + eps[(size_t)m * AD + n] * __expf(0.5f * lv);
                size_t o = (size_t)m * AD + n;
                out_mean[o] = mean; out_lv[o] = lv; out_a[o] = av; a16[o] = (_Float16)av;
            }
        }
    }
}

// ---------------- sequential LSTM + softmax(alpha): one persistent workgroup ----------------
__global__ void lstm_alpha(const float* __restrict__ a, const float* __restrict__ a_init,
                           const float* __restrict__ W_ih, const float* __restrict__ W_hh,
                           const float* __restrict__ b_ih, const float* __restrict__ b_hh,
                           const float* __restrict__ Wa, const float* __restrict__ ba,
                           float* __restrict__ alpha) {
    __shared__ float sh[BB * HH], sc[BB * HH], sg[BB * 4 * HH], sa[BB * AD], slog[BB * KMIX];
    int tid = threadIdx.x;
    for (int i = tid; i < BB * HH; i += blockDim.x) { sh[i] = 0.f; sc[i] = 0.f; }
    __syncthreads();
    for (int t = 0; t < SEQ; ++t) {
        for (int i = tid; i < BB * AD; i += blockDim.x) {
            int b = i >> 3, k = i & 7;
            sa[i] = (t == 0) ? a_init[k] : a[(((size_t)(t - 1) * BB + b)) * AD + k];
        }
        __syncthreads();
        for (int i = tid; i < BB * 4 * HH; i += blockDim.x) {
            int b = i / (4 * HH), j = i - b * 4 * HH;
            float g = b_ih[j] + b_hh[j];
            for (int k = 0; k < AD; ++k) g += sa[b * AD + k] * W_ih[j * AD + k];
            for (int k = 0; k < HH; ++k) g += sh[b * HH + k] * W_hh[j * HH + k];
            sg[i] = g;
        }
        __syncthreads();
        for (int i = tid; i < BB * HH; i += blockDim.x) {
            int b = i / HH, k = i - b * HH;
            float ig = sigmoidf_(sg[b * 4 * HH + k]);
            float fg = sigmoidf_(sg[b * 4 * HH + HH + k]);
            float gg = tanhf(sg[b * 4 * HH + 2 * HH + k]);
            float og = sigmoidf_(sg[b * 4 * HH + 3 * HH + k]);
            float cc = fg * sc[i] + ig * gg;
            sc[i] = cc; sh[i] = og * tanhf(cc);
        }
        __syncthreads();
        for (int i = tid; i < BB * KMIX; i += blockDim.x) {
            int b = i / KMIX, kk = i - b * KMIX;
            float l = ba[kk];
            for (int k = 0; k < HH; ++k) l += sh[b * HH + k] * Wa[kk * HH + k];
            slog[i] = l;
        }
        __syncthreads();
        for (int i = tid; i < BB; i += blockDim.x) {
            float m = fmaxf(slog[i * 3], fmaxf(slog[i * 3 + 1], slog[i * 3 + 2]));
            float e0 = __expf(slog[i * 3] - m), e1 = __expf(slog[i * 3 + 1] - m), e2 = __expf(slog[i * 3 + 2] - m);
            float s = e0 + e1 + e2;
            size_t o = ((size_t)t * BB + i) * KMIX;
            alpha[o] = e0 / s; alpha[o + 1] = e1 / s; alpha[o + 2] = e2 / s;
        }
        __syncthreads();
    }
}

// ---------------- Kalman filter: one wave, lane = batch element ----------------
__global__ void kalman_filter(const float* __restrict__ a, const float* __restrict__ a_init,
                              const float* __restrict__ alpha,
                              const float* __restrict__ Amats, const float* __restrict__ Bmats,
                              const float* __restrict__ Cmats,
                              float* __restrict__ mu_pred, float* __restrict__ mu_filt,
                              float* __restrict__ Sig_pred, float* __restrict__ Sig_filt,
                              float* __restrict__ mu_sm, float* __restrict__ Sig_sm) {
    int b = threadIdx.x;
    if (b >= BB) return;
    float mu[ZD] = {0, 0, 0, 0};
    float Sig[ZD * ZD];
    for (int i = 0; i < 16; ++i) Sig[i] = (i % 5 == 0) ? 20.f : 0.f;
    for (int t = 0; t < SEQ; ++t) {
        float al0 = alpha[((size_t)t * BB + b) * 3 + 0];
        float al1 = alpha[((size_t)t * BB + b) * 3 + 1];
        float al2 = alpha[((size_t)t * BB + b) * 3 + 2];
        float Am[16], Bm_[32], Cm_[32];
        for (int i = 0; i < 16; ++i) Am[i]  = al0 * Amats[i] + al1 * Amats[16 + i] + al2 * Amats[32 + i];
        for (int i = 0; i < 32; ++i) Bm_[i] = al0 * Bmats[i] + al1 * Bmats[32 + i] + al2 * Bmats[64 + i];
        for (int i = 0; i < 32; ++i) Cm_[i] = al0 * Cmats[i] + al1 * Cmats[32 + i] + al2 * Cmats[64 + i];
        float u[AD], at[AD];
        for (int j = 0; j < AD; ++j) {
            u[j] = (t == 0) ? a_init[j] : a[((size_t)(t - 1) * BB + b) * AD + j];
            at[j] = a[((size_t)t * BB + b) * AD + j];
        }
        float mup[ZD], Sp[16];
        if (t == 0) {
            for (int i = 0; i < ZD; ++i) mup[i] = 0.f;
            for (int i = 0; i < 16; ++i) Sp[i] = (i % 5 == 0) ? 20.f : 0.f;
        } else {
            for (int i = 0; i < ZD; ++i) {
                float s = 0.f;
                for (int j = 0; j < ZD; ++j) s += Am[i * 4 + j] * mu[j];
                for (int j = 0; j < AD; ++j) s += Bm_[i * 8 + j] * u[j];
                mup[i] = s;
            }
            float T[16];
            for (int i = 0; i < 4; ++i) for (int j = 0; j < 4; ++j) {
                float s = 0.f;
                for (int k = 0; k < 4; ++k) s += Am[i * 4 + k] * Sig[k * 4 + j];
                T[i * 4 + j] = s;
            }
            for (int i = 0; i < 4; ++i) for (int j = 0; j < 4; ++j) {
                float s = 0.f;
                for (int k = 0; k < 4; ++k) s += T[i * 4 + k] * Am[j * 4 + k];
                Sp[i * 4 + j] = s + ((i == j) ? 0.08f : 0.f);
            }
        }
        float res[AD];
        for (int i = 0; i < AD; ++i) {
            float s = 0.f;
            for (int j = 0; j < ZD; ++j) s += Cm_[i * 4 + j] * mup[j];
            res[i] = at[i] - s;
        }
        float CS[32];
        for (int i = 0; i < 8; ++i) for (int j = 0; j < 4; ++j) {
            float s = 0.f;
            for (int k = 0; k < 4; ++k) s += Cm_[i * 4 + k] * Sp[k * 4 + j];
            CS[i * 4 + j] = s;
        }
        float M[8][16];
        for (int i = 0; i < 8; ++i) {
            for (int j = 0; j < 8; ++j) {
                float s = 0.f;
                for (int k = 0; k < 4; ++k) s += CS[i * 4 + k] * Cm_[j * 4 + k];
                M[i][j] = s + ((i == j) ? 0.03f : 0.f);
                M[i][8 + j] = (i == j) ? 1.f : 0.f;
            }
        }
        for (int p = 0; p < 8; ++p) {
            float inv = 1.f / M[p][p];
            for (int j = 0; j < 16; ++j) M[p][j] *= inv;
            for (int i = 0; i < 8; ++i) if (i != p) {
                float f = M[i][p];
                for (int j = 0; j < 16; ++j) M[i][j] -= f * M[p][j];
            }
        }
        float PC[32];
        for (int i = 0; i < 4; ++i) for (int j = 0; j < 8; ++j) {
            float s = 0.f;
            for (int k = 0; k < 4; ++k) s += Sp[i * 4 + k] * Cm_[j * 4 + k];
            PC[i * 8 + j] = s;
        }
        float Kt[32];
        for (int i = 0; i < 4; ++i) for (int j = 0; j < 8; ++j) {
            float s = 0.f;
            for (int k = 0; k < 8; ++k) s += PC[i * 8 + k] * M[k][8 + j];
            Kt[i * 8 + j] = s;
        }
        float muf[ZD];
        for (int i = 0; i < ZD; ++i) {
            float s = mup[i];
            for (int j = 0; j < 8; ++j) s += Kt[i * 8 + j] * res[j];
            muf[i] = s;
        }
        float IKC[16];
        for (int i = 0; i < 4; ++i) for (int j = 0; j < 4; ++j) {
            float s = 0.f;
            for (int k = 0; k < 8; ++k) s += Kt[i * 8 + k] * Cm_[k * 4 + j];
            IKC[i * 4 + j] = ((i == j) ? 1.f : 0.f) - s;
        }
        float T2[16];
        for (int i = 0; i < 4; ++i) for (int j = 0; j < 4; ++j) {
            float s = 0.f;
            for (int k = 0; k < 4; ++k) s += IKC[i * 4 + k] * Sp[k * 4 + j];
            T2[i * 4 + j] = s;
        }
        float Sf[16];
        for (int i = 0; i < 4; ++i) for (int j = 0; j < 4; ++j) {
            float s = 0.f;
            for (int k = 0; k < 4; ++k) s += T2[i * 4 + k] * IKC[j * 4 + k];
            float kkt = 0.f;
            for (int k = 0; k < 8; ++k) kkt += Kt[i * 8 + k] * Kt[j * 8 + k];
            Sf[i * 4 + j] = s + 0.03f * kkt;
        }
        size_t ob = (size_t)t * BB + b;
        for (int i = 0; i < ZD; ++i) { mu_pred[ob * 4 + i] = mup[i]; mu_filt[ob * 4 + i] = muf[i]; }
        for (int i = 0; i < 16; ++i) { Sig_pred[ob * 16 + i] = Sp[i]; Sig_filt[ob * 16 + i] = Sf[i]; }
        for (int i = 0; i < ZD; ++i) mu[i] = muf[i];
        for (int i = 0; i < 16; ++i) Sig[i] = Sf[i];
    }
    size_t last = (size_t)(SEQ - 1) * BB + b;
    for (int i = 0; i < ZD; ++i) mu_sm[last * 4 + i] = mu[i];
    for (int i = 0; i < 16; ++i) Sig_sm[last * 16 + i] = Sig[i];
}

// ---------------- RTS smoother (backward): one wave, lane = batch element ----------------
__global__ void rts_smoother(const float* __restrict__ alpha, const float* __restrict__ Amats,
                             const float* __restrict__ mu_filt, const float* __restrict__ Sig_filt,
                             const float* __restrict__ Sig_pred,
                             float* __restrict__ mu_sm, float* __restrict__ Sig_sm) {
    int b = threadIdx.x;
    if (b >= BB) return;
    float mus[ZD], Ss[16];
    size_t last = (size_t)(SEQ - 1) * BB + b;
    for (int i = 0; i < ZD; ++i) mus[i] = mu_filt[last * 4 + i];
    for (int i = 0; i < 16; ++i) Ss[i] = Sig_filt[last * 16 + i];
    for (int t = SEQ - 2; t >= 0; --t) {
        size_t cur = (size_t)t * BB + b, nxt = (size_t)(t + 1) * BB + b;
        float al0 = alpha[nxt * 3 + 0], al1 = alpha[nxt * 3 + 1], al2 = alpha[nxt * 3 + 2];
        float An[16];
        for (int i = 0; i < 16; ++i) An[i] = al0 * Amats[i] + al1 * Amats[16 + i] + al2 * Amats[32 + i];
        float Sf[16], Spn[16], muf[ZD], mufn[ZD];
        for (int i = 0; i < 16; ++i) { Sf[i] = Sig_filt[cur * 16 + i]; Spn[i] = Sig_pred[nxt * 16 + i]; }
        for (int i = 0; i < ZD; ++i) { muf[i] = mu_filt[cur * 4 + i]; mufn[i] = mu_filt[nxt * 4 + i]; }
        float M[4][8];
        for (int i = 0; i < 4; ++i)
            for (int j = 0; j < 4; ++j) { M[i][j] = Spn[i * 4 + j]; M[i][4 + j] = (i == j) ? 1.f : 0.f; }
        for (int p = 0; p < 4; ++p) {
            float inv = 1.f / M[p][p];
            for (int j = 0; j < 8; ++j) M[p][j] *= inv;
            for (int i = 0; i < 4; ++i) if (i != p) {
                float f = M[i][p];
                for (int j = 0; j < 8; ++j) M[i][j] -= f * M[p][j];
            }
        }
        float T[16], J[16];
        for (int i = 0; i < 4; ++i) for (int j = 0; j < 4; ++j) {
            float s = 0.f;
            for (int k = 0; k < 4; ++k) s += Sf[i * 4 + k] * An[j * 4 + k];
            T[i * 4 + j] = s;
        }
        for (int i = 0; i < 4; ++i) for (int j = 0; j < 4; ++j) {
            float s = 0.f;
            for (int k = 0; k < 4; ++k) s += T[i * 4 + k] * M[k][4 + j];
            J[i * 4 + j] = s;
        }
        float mun[ZD];
        for (int i = 0; i < ZD; ++i) {
            float s = muf[i];
            for (int j = 0; j < ZD; ++j) s += J[i * 4 + j] * (mus[j] - mufn[j]);
            mun[i] = s;
        }
        float D[16], T2[16], Sn[16];
        for (int i = 0; i < 16; ++i) D[i] = Ss[i] - Spn[i];
        for (int i = 0; i < 4; ++i) for (int j = 0; j < 4; ++j) {
            float s = 0.f;
            for (int k = 0; k < 4; ++k) s += J[i * 4 + k] * D[k * 4 + j];
            T2[i * 4 + j] = s;
        }
        for (int i = 0; i < 4; ++i) for (int j = 0; j < 4; ++j) {
            float s = 0.f;
            for (int k = 0; k < 4; ++k) s += T2[i * 4 + k] * J[j * 4 + k];
            Sn[i * 4 + j] = Sf[i * 4 + j] + s;
        }
        for (int i = 0; i < ZD; ++i) mu_sm[cur * 4 + i] = mun[i];
        for (int i = 0; i < 16; ++i) Sig_sm[cur * 16 + i] = Sn[i];
        for (int i = 0; i < ZD; ++i) mus[i] = mun[i];
        for (int i = 0; i < 16; ++i) Ss[i] = Sn[i];
    }
}

// ---------------- host orchestration ----------------
extern "C" void kernel_launch(void* const* d_in, const int* in_sizes, int n_in,
                              void* d_out, int out_size, void* d_ws, size_t ws_size,
                              hipStream_t stream) {
    (void)in_sizes; (void)n_in; (void)out_size; (void)ws_size;
    const float* x     = (const float*)d_in[0];
    const float* eps   = (const float*)d_in[1];
    const float* Wc0   = (const float*)d_in[2];   const float* bc0 = (const float*)d_in[3];
    const float* Wc1   = (const float*)d_in[4];   const float* bc1 = (const float*)d_in[5];
    const float* Wc2   = (const float*)d_in[6];   const float* bc2 = (const float*)d_in[7];
    const float* Wm    = (const float*)d_in[8];   const float* bm  = (const float*)d_in[9];
    const float* Wlv   = (const float*)d_in[10];  const float* blv = (const float*)d_in[11];
    const float* Wd    = (const float*)d_in[12];  const float* bd  = (const float*)d_in[13];
    const float* Wt0   = (const float*)d_in[14];  const float* bt0 = (const float*)d_in[15];
    const float* Wt1   = (const float*)d_in[16];  const float* bt1 = (const float*)d_in[17];
    const float* Wt2   = (const float*)d_in[18];  const float* bt2 = (const float*)d_in[19];
    const float* Wg    = (const float*)d_in[20];  const float* bg  = (const float*)d_in[21];
    const float* Amats = (const float*)d_in[22];
    const float* Bmats = (const float*)d_in[23];
    const float* Cmats = (const float*)d_in[24];
    const float* a_init= (const float*)d_in[25];
    const float* W_ih  = (const float*)d_in[26];  const float* W_hh = (const float*)d_in[27];
    const float* b_ih  = (const float*)d_in[28];  const float* b_hh = (const float*)d_in[29];
    const float* Wa    = (const float*)d_in[30];  const float* ba   = (const float*)d_in[31];

    float* out = (float*)d_out;
    float* out_logy  = out;                               // (S,B,1024)
    float* out_a     = out_logy + (size_t)SBN * XD;       // (S,B,8)
    float* out_mean  = out_a + (size_t)SBN * AD;
    float* out_lv    = out_mean + (size_t)SBN * AD;
    float* out_mus   = out_lv + (size_t)SBN * AD;         // (S,B,4)
    float* out_Sigs  = out_mus + (size_t)SBN * ZD;        // (S,B,4,4)

    char* w = (char*)d_ws;
    size_t off = 0;
    auto a16p = [&](size_t n) { _Float16* p = (_Float16*)(w + off); off += ((n * 2 + 255) / 256) * 256; return p; };
    auto a32p = [&](size_t n) { float* p = (float*)(w + off); off += ((n * 4 + 255) / 256) * 256; return p; };

    _Float16* x16  = a16p((size_t)SBN * XD);              // (SB,1,32,32)
    _Float16* h1   = a16p((size_t)SBN * 32 * 15 * 15);
    _Float16* h2   = a16p((size_t)SBN * 32 * 7 * 7);
    _Float16* feat = a16p((size_t)SBN * CN);
    _Float16* a16  = a16p((size_t)SBN * AD);
    _Float16* d0   = a16p((size_t)SBN * CN);
    _Float16* dt0  = a16p((size_t)SBN * 32 * 7 * 7);
    _Float16* dt1  = a16p((size_t)SBN * 32 * 15 * 15);
    _Float16* pWc0 = a16p(32 * 32);
    _Float16* pWc1 = a16p(288 * 32);
    _Float16* pWc2 = a16p(288 * 32);
    _Float16* pWt0 = a16p(288 * 32);
    _Float16* pWt1 = a16p(288 * 32);
    _Float16* pWt2 = a16p(288 * 32);
    _Float16* pWm  = a16p(288 * 16);
    _Float16* pWlv = a16p(288 * 16);
    _Float16* pWd  = a16p(32 * 288);
    float* alpha    = a32p((size_t)SEQ * BB * KMIX);
    float* mu_pred  = a32p((size_t)SEQ * BB * ZD);
    float* mu_filt  = a32p((size_t)SEQ * BB * ZD);
    float* Sig_pred = a32p((size_t)SEQ * BB * 16);
    float* Sig_filt = a32p((size_t)SEQ * BB * 16);

    // --- weight packing + x conversion ---
    {
        int n = SBN * XD;
        f32_to_f16_kernel<<<(n + 255) / 256, 256, 0, stream>>>(x, x16, n);
        pack_conv_w<<<(32 * 32 + 255) / 256, 256, 0, stream>>>(Wc0, pWc0, 1, 32, 32, 0);
        pack_conv_w<<<(288 * 32 + 255) / 256, 256, 0, stream>>>(Wc1, pWc1, 32, 32, 288, 0);
        pack_conv_w<<<(288 * 32 + 255) / 256, 256, 0, stream>>>(Wc2, pWc2, 32, 32, 288, 0);
        pack_conv_w<<<(288 * 32 + 255) / 256, 256, 0, stream>>>(Wt0, pWt0, 32, 32, 288, 1);
        pack_conv_w<<<(288 * 32 + 255) / 256, 256, 0, stream>>>(Wt1, pWt1, 32, 32, 288, 1);
        pack_conv_w<<<(288 * 32 + 255) / 256, 256, 0, stream>>>(Wt2, pWt2, 32, 32, 288, 1);
        pack_mat_T<<<(288 * 16 + 255) / 256, 256, 0, stream>>>(Wm, pWm, 8, 288, 288, 16);
        pack_mat_T<<<(288 * 16 + 255) / 256, 256, 0, stream>>>(Wlv, pWlv, 8, 288, 288, 16);
        pack_mat_T<<<(32 * 288 + 255) / 256, 256, 0, stream>>>(Wd, pWd, 288, 8, 32, 288);
    }

    auto grid_for = [](int Mtot) { int mt = (Mtot + 15) / 16; return dim3((mt + 3) / 4); };
    dim3 blk(128);
    auto smem = [](int Kp, int Co) { return (size_t)Kp * Co * 2; };

    // --- encoder ---
    conv_gemm<<<grid_for(SBN * 15 * 15), blk, smem(32, 32), stream>>>(x16, pWc0, bc0, h1, nullptr, nullptr, nullptr,
        SBN, 1, 32, 32, 15, 15, 32, 9, 32, 0, 1);
    conv_gemm<<<grid_for(SBN * 7 * 7), blk, smem(288, 32), stream>>>(h1, pWc1, bc1, h2, nullptr, nullptr, nullptr,
        SBN, 32, 15, 15, 7, 7, 32, 288, 288, 0, 1);
    conv_gemm<<<grid_for(SBN * 3 * 3), blk, smem(288, 32), stream>>>(h2, pWc2, bc2, feat, nullptr, nullptr, nullptr,
        SBN, 32, 7, 7, 3, 3, 32, 288, 288, 0, 1);
    encode_head<<<grid_for(SBN), blk, 0, stream>>>(feat, pWm, pWlv, bm, blv, eps,
        out_a, out_mean, out_lv, a16);

    // --- decoder ---
    conv_gemm<<<grid_for(SBN), blk, smem(32, 288), stream>>>(a16, pWd, bd, d0, nullptr, nullptr, nullptr,
        SBN, 0, 1, 1, 1, 1, 288, 8, 32, 2, 0);
    conv_gemm<<<grid_for(SBN * 7 * 7), blk, smem(288, 32), stream>>>(d0, pWt0, bt0, dt0, nullptr, nullptr, nullptr,
        SBN, 32, 3, 3, 7, 7, 32, 288, 288, 1, 1);
    conv_gemm<<<grid_for(SBN * 15 * 15), blk, smem(288, 32), stream>>>(dt0, pWt1, bt1, dt1, nullptr, nullptr, nullptr,
        SBN, 32, 7, 7, 15, 15, 32, 288, 288, 1, 1);
    // convT2 fused with relu + 1x1 gate + sigmoid -> log_y
    conv_gemm<<<grid_for(SBN * 32 * 32), blk, smem(288, 32), stream>>>(dt1, pWt2, bt2, nullptr, out_logy, Wg, bg,
        SBN, 32, 15, 15, 32, 32, 32, 288, 288, 1, 1);

    // --- sequential state-space part ---
    lstm_alpha<<<1, 256, 0, stream>>>(out_a, a_init, W_ih, W_hh, b_ih, b_hh, Wa, ba, alpha);
    kalman_filter<<<1, 32, 0, stream>>>(out_a, a_init, alpha, Amats, Bmats, Cmats,
        mu_pred, mu_filt, Sig_pred, Sig_filt, out_mus, out_Sigs);
    rts_smoother<<<1, 32, 0, stream>>>(alpha, Amats, mu_filt, Sig_filt, Sig_pred, out_mus, out_Sigs);
}
